// MoEDecoderLayer_48627619726051
// MI455X (gfx1250) — matrix operand order repaired and checked by reading, not compiled, not run on hardware
//
#include <hip/hip_runtime.h>
#include <hip/hip_bf16.h>

// MoE decoder layer for gfx1250 (MI455X).
// - All large GEMMs: v_wmma_f32_16x16x32_bf16, fp32 accumulation.
// - Weights pre-converted once per launch to bf16 (and pre-transposed to
//   [N,K] for gate/up/down) so GEMM tile fills are contiguous 16B copies.
// - GEMM tile fills use async global->LDS copies (ASYNCcnt) with double
//   buffering: s_wait_asynccnt / barrier / issue-next / compute-current.
// - Dense projections + down-projection use 128-row M tiles: 8 WMMAs per
//   wave per K-tile (2x compute per synchronization point).
// - Top-1 routing computed exactly; only the routed expert is evaluated
//   (identical math to the reference's dense compute + one-hot select).

typedef __bf16 bf16;
typedef __attribute__((ext_vector_type(16))) __bf16 v16bf;
typedef __attribute__((ext_vector_type(8)))  float  v8f;

#define B_  2
#define S_  2048
#define D_  1024
#define H_  16
#define DH  64
#define F_  4096
#define E_  4
#define T_  (B_ * S_)   // 4096 tokens

// ---------------------------------------------------------------- fragments
union FragBF { v16bf v; uint4 u[2]; };

// A-matrix 16x32 bf16 fragment (row-major source, stride in elements).
__device__ inline v16bf load_a_frag(const bf16* base, int stride) {
  const int lane = threadIdx.x & 31;
  const int row = lane & 15, kg = lane >> 4;
  const bf16* p = base + row * stride + kg * 8;
  FragBF f;
  f.u[0] = *(const uint4*)(p);        // K = kg*8 + 0..7      -> v0..v3
  f.u[1] = *(const uint4*)(p + 16);   // K = 16 + kg*8 + 0..7 -> v4..v7
  return f.v;
}

// B-matrix 32x16 bf16 fragment from column-major storage [n][k].
__device__ inline v16bf load_b_frag(const bf16* base, int stride) {
  const int lane = threadIdx.x & 31;
  const int col = lane & 15, kg = lane >> 4;
  const bf16* p = base + col * stride + kg * 8;
  FragBF f;
  f.u[0] = *(const uint4*)(p);        // K = kg*8 + 0..7
  f.u[1] = *(const uint4*)(p + 16);   // K = 16 + kg*8 + 0..7
  return f.v;
}

__device__ inline v8f wmma_bf16(v16bf a, v16bf b, v8f c) {
  return __builtin_amdgcn_wmma_f32_16x16x32_bf16(
      false, a, false, b, (short)0, c, false, false);
}

// ------------------------------------------------- async global->LDS copy
// Per-lane 16B copy tracked by ASYNCcnt. LDS byte offset = low 32 bits of
// the generic shared-memory address (flat LDS aperture: addr[31:0]).
__device__ inline void async_b128(const bf16* lds_p, const bf16* g) {
  unsigned ldsoff = (unsigned)(unsigned long long)lds_p;
  asm volatile("global_load_async_to_lds_b128 %0, %1, off"
               :: "v"(ldsoff), "v"(g)
               : "memory");
}
__device__ inline void wait_async0() {
  asm volatile("s_wait_asynccnt 0x0" ::: "memory");
}

// ---------------------------------------------------------------- rmsnorm
__global__ __launch_bounds__(256)
void rmsnorm_bf16(const float* __restrict__ x, const float* __restrict__ w,
                  bf16* __restrict__ out) {
  __shared__ float red[8];
  const int row = blockIdx.x;
  const float* xr = x + (size_t)row * D_;
  float ss = 0.f;
  for (int j = threadIdx.x; j < D_; j += 256) { float v = xr[j]; ss += v * v; }
  for (int o = 16; o > 0; o >>= 1) ss += __shfl_xor(ss, o, 32);
  if ((threadIdx.x & 31) == 0) red[threadIdx.x >> 5] = ss;
  __syncthreads();
  float tot = 0.f;
#pragma unroll
  for (int i = 0; i < 8; ++i) tot += red[i];
  const float rinv = rsqrtf(tot / (float)D_ + 1e-6f);
  bf16* orow = out + (size_t)row * D_;
  for (int j = threadIdx.x; j < D_; j += 256)
    orow[j] = (bf16)(xr[j] * rinv * w[j]);
}

// ------------------------------------------------ weight pre-conversion
__global__ __launch_bounds__(256)
void convert_f32_bf16(const float* __restrict__ in, bf16* __restrict__ out,
                      int n) {
  int i = (blockIdx.x * 256 + threadIdx.x) * 4;
  if (i < n) {
    float4 v = *(const float4*)&in[i];
    bf16* o = &out[i];
    o[0] = (bf16)v.x; o[1] = (bf16)v.y; o[2] = (bf16)v.z; o[3] = (bf16)v.w;
  }
}

// in: fp32 [K,N] (expert blockIdx.z) -> out: bf16 [N,K]
__global__ __launch_bounds__(256)
void transpose_f32_bf16(const float* __restrict__ in, bf16* __restrict__ out,
                        int K, int N) {
  __shared__ bf16 Ls[64 * 65];
  const size_t ei = (size_t)blockIdx.z * K * N;
  const int k0 = blockIdx.y * 64, n0 = blockIdx.x * 64;
  for (int i = threadIdx.x; i < 4096; i += 256) {
    int kk = i >> 6, n = i & 63;                       // coalesced over n
    Ls[n * 65 + kk] = (bf16)in[ei + (size_t)(k0 + kk) * N + n0 + n];
  }
  __syncthreads();
  for (int i = threadIdx.x; i < 4096; i += 256) {
    int n = i >> 6, kk = i & 63;                       // coalesced over kk
    out[ei + (size_t)(n0 + n) * K + k0 + kk] = Ls[n * 65 + kk];
  }
}

// ------------- C = A @ B^T, async double-buffered, 128x64 tile (bf16 ops)
// A: bf16 [M,K] row-major. Wb: bf16 [N,K] row-major. EPI: 0=store, 1=+resid.
template <int EPI>
__global__ __launch_bounds__(256)
void gemm_nt_async(const bf16* __restrict__ A, const bf16* __restrict__ Wb,
                   const float* __restrict__ resid, float* __restrict__ out,
                   int M, int N, int K) {
  __shared__ alignas(16) bf16 As[2][128 * 72];   // 36864 B
  __shared__ alignas(16) bf16 Bs[2][64 * 72];    // 18432 B
  const int m0 = blockIdx.y * 128, n0 = blockIdx.x * 64;
  const int tid = threadIdx.x, w = tid >> 5, lane = tid & 31;
  const int wm = w >> 2;   // 0..1 : 64-row band (4 x 16-row C tiles)
  const int wn = w & 3;    // 0..3 : 16-col band

  auto issue = [&](int k0, int buf) {
#pragma unroll
    for (int i = 0; i < 4; ++i) {                // A: 128x64 bf16
      int idx = tid + i * 256, r = idx >> 3, q = (idx & 7) * 8;
      async_b128(&As[buf][r * 72 + q], &A[(size_t)(m0 + r) * K + k0 + q]);
    }
#pragma unroll
    for (int i = 0; i < 2; ++i) {                // B: 64x64 bf16
      int idx = tid + i * 256, r = idx >> 3, q = (idx & 7) * 8;
      async_b128(&Bs[buf][r * 72 + q], &Wb[(size_t)(n0 + r) * K + k0 + q]);
    }
  };

  issue(0, 0);
  v8f c[4] = {};
  const int nk = K >> 6;
  for (int t = 0; t < nk; ++t) {
    const int buf = t & 1;
    wait_async0();
    __syncthreads();
    if (t + 1 < nk) issue((t + 1) << 6, buf ^ 1);
#pragma unroll
    for (int kc = 0; kc < 2; ++kc) {
      v16bf bfr = load_b_frag(&Bs[buf][(wn * 16) * 72 + kc * 32], 72);
#pragma unroll
      for (int ms = 0; ms < 4; ++ms) {
        v16bf a = load_a_frag(&As[buf][(wm * 64 + ms * 16) * 72 + kc * 32], 72);
        c[ms] = wmma_bf16(a, bfr, c[ms]);
      }
    }
  }
  const int cn = n0 + wn * 16 + (lane & 15);
#pragma unroll
  for (int ms = 0; ms < 4; ++ms) {
    const int rb = m0 + wm * 64 + ms * 16 + ((lane >> 4) << 3);
#pragma unroll
    for (int r = 0; r < 8; ++r) {
      float v = c[ms][r];
      if (EPI == 1) v += resid[(size_t)(rb + r) * N + cn];
      out[(size_t)(rb + r) * N + cn] = v;
    }
  }
}

// --------------------------------------------------- RoPE + [B,H,S,Dh] relayout
__global__ __launch_bounds__(256)
void rope_relayout(const float* __restrict__ q, const float* __restrict__ k,
                   const float* __restrict__ v, bf16* __restrict__ qb,
                   bf16* __restrict__ kb, bf16* __restrict__ vb) {
  const int tok = blockIdx.x;
  const int b = tok / S_, s = tok % S_;
  const size_t base = (size_t)tok * D_;
  for (int i = threadIdx.x; i < D_; i += 256) {
    int h = i >> 6, dh = i & 63;
    float inv = __powf(10000.f, -(float)(dh & 31) / 32.f);
    float sn, cs;
    __sincosf((float)s * inv, &sn, &cs);
    float qv = q[base + i], kv = k[base + i];
    float qp, kp;
    if (dh < 32) { qp = -q[base + i + 32]; kp = -k[base + i + 32]; }
    else         { qp =  q[base + i - 32]; kp =  k[base + i - 32]; }
    size_t o = ((size_t)(b * H_ + h) * S_ + s) * DH + dh;
    qb[o] = (bf16)(qv * cs + qp * sn);
    kb[o] = (bf16)(kv * cs + kp * sn);
    vb[o] = (bf16)v[base + i];
  }
}

// ---------------------------------------------------------- flash attention
// One block = (b, h, 64 Q rows); 4 waves, each owns 16 Q rows end-to-end.
__global__ __launch_bounds__(128)
void flash_attn(const bf16* __restrict__ qb, const bf16* __restrict__ kb,
                const bf16* __restrict__ vb, bf16* __restrict__ ctx) {
  __shared__ alignas(16) bf16 Ps[4][16 * 72];   // per-wave P tile (16 x 64)
  __shared__ alignas(16) bf16 VTs[64 * 72];     // V^T tile: [dh][key]
  const int b = blockIdx.z, h = blockIdx.y, q0 = blockIdx.x * 64;
  const int tid = threadIdx.x, w = tid >> 5, lane = tid & 31;
  const int akg = lane >> 4;
  const bf16* Qb = qb + (size_t)(b * H_ + h) * S_ * DH;
  const bf16* Kb = kb + (size_t)(b * H_ + h) * S_ * DH;
  const bf16* Vb = vb + (size_t)(b * H_ + h) * S_ * DH;

  const int arow = q0 + w * 16 + (lane & 15);
  v16bf qf[2];
#pragma unroll
  for (int kc = 0; kc < 2; ++kc) {
    FragBF f;
    const bf16* p = Qb + (size_t)arow * DH + kc * 32 + akg * 8;
    f.u[0] = *(const uint4*)p;
    f.u[1] = *(const uint4*)(p + 16);
    qf[kc] = f.v;
  }

  v8f o[4] = {};
  float m_[8], l_[8];
#pragma unroll
  for (int r = 0; r < 8; ++r) { m_[r] = -1e30f; l_[r] = 0.f; }

  const int qrow0 = q0 + w * 16 + ((lane >> 4) << 3);

  for (int k0 = 0; k0 <= q0; k0 += 64) {
    __syncthreads();
    for (int i = tid; i < 64 * 64; i += 128) {     // V^T into LDS
      int key = i >> 6, dh = i & 63;
      VTs[dh * 72 + key] = Vb[(size_t)(k0 + key) * DH + dh];
    }

    v8f sc[4];
#pragma unroll
    for (int nt = 0; nt < 4; ++nt) {
      v8f acc = {};
#pragma unroll
      for (int kc = 0; kc < 2; ++kc) {
        FragBF f;
        const bf16* p =
            Kb + (size_t)(k0 + nt * 16 + (lane & 15)) * DH + kc * 32 + akg * 8;
        f.u[0] = *(const uint4*)p;
        f.u[1] = *(const uint4*)(p + 16);
        acc = wmma_bf16(qf[kc], f.v, acc);
      }
      sc[nt] = acc;
    }

#pragma unroll
    for (int nt = 0; nt < 4; ++nt) {
      int kcol = k0 + nt * 16 + (lane & 15);
#pragma unroll
      for (int r = 0; r < 8; ++r) {
        float sv = sc[nt][r] * 0.125f;             // 1/sqrt(64)
        if (kcol > qrow0 + r) sv = -1e30f;
        sc[nt][r] = sv;
      }
    }

#pragma unroll
    for (int r = 0; r < 8; ++r) {
      float rm = fmaxf(fmaxf(sc[0][r], sc[1][r]), fmaxf(sc[2][r], sc[3][r]));
      rm = fmaxf(rm, __shfl_xor(rm, 1, 32));
      rm = fmaxf(rm, __shfl_xor(rm, 2, 32));
      rm = fmaxf(rm, __shfl_xor(rm, 4, 32));
      rm = fmaxf(rm, __shfl_xor(rm, 8, 32));
      float mnew = fmaxf(m_[r], rm);
      float fsc = __expf(m_[r] - mnew);
      float rs = 0.f;
#pragma unroll
      for (int nt = 0; nt < 4; ++nt) {
        float p = __expf(sc[nt][r] - mnew);
        sc[nt][r] = p;
        rs += p;
      }
      rs += __shfl_xor(rs, 1, 32);
      rs += __shfl_xor(rs, 2, 32);
      rs += __shfl_xor(rs, 4, 32);
      rs += __shfl_xor(rs, 8, 32);
      l_[r] = l_[r] * fsc + rs;
      m_[r] = mnew;
#pragma unroll
      for (int nt = 0; nt < 4; ++nt) o[nt][r] *= fsc;
    }

    bf16* Pw = &Ps[w][0];
#pragma unroll
    for (int nt = 0; nt < 4; ++nt) {
      int colp = nt * 16 + (lane & 15);
#pragma unroll
      for (int r = 0; r < 8; ++r)
        Pw[(((lane >> 4) << 3) + r) * 72 + colp] = (bf16)sc[nt][r];
    }
    __syncthreads();

#pragma unroll
    for (int kc = 0; kc < 2; ++kc) {
      FragBF pa;
      const bf16* pp = Pw + (lane & 15) * 72 + kc * 32 + akg * 8;
      pa.u[0] = *(const uint4*)pp;
      pa.u[1] = *(const uint4*)(pp + 16);
#pragma unroll
      for (int nt = 0; nt < 4; ++nt) {
        FragBF vbf;
        const bf16* vp = &VTs[(nt * 16 + (lane & 15)) * 72 + kc * 32 + akg * 8];
        vbf.u[0] = *(const uint4*)vp;
        vbf.u[1] = *(const uint4*)(vp + 16);
        o[nt] = wmma_bf16(pa.v, vbf.v, o[nt]);
      }
    }
  }

#pragma unroll
  for (int nt = 0; nt < 4; ++nt) {
#pragma unroll
    for (int r = 0; r < 8; ++r) {
      int s = qrow0 + r;
      int dh = nt * 16 + (lane & 15);
      ctx[(size_t)(b * S_ + s) * D_ + h * 64 + dh] = (bf16)(o[nt][r] / l_[r]);
    }
  }
}

// ---------------------------------------------------------------- router
__global__ __launch_bounds__(256)
void router_kernel(const bf16* __restrict__ x2, const float* __restrict__ rw,
                   const float* __restrict__ rb, int* __restrict__ cnt,
                   int* __restrict__ list, float* __restrict__ tot_prob,
                   float* __restrict__ zsum) {
  const int w = threadIdx.x >> 5, lane = threadIdx.x & 31;
  const int tok = blockIdx.x * 8 + w;
  const bf16* xr = x2 + (size_t)tok * D_;
  float acc[E_] = {0.f, 0.f, 0.f, 0.f};
  for (int j = lane; j < D_; j += 32) {
    float xv = (float)xr[j];
#pragma unroll
    for (int e = 0; e < E_; ++e) acc[e] += xv * rw[e * D_ + j];
  }
#pragma unroll
  for (int e = 0; e < E_; ++e)
    for (int o = 16; o > 0; o >>= 1) acc[e] += __shfl_xor(acc[e], o, 32);
  if (lane == 0) {
    float lg[E_], mx = -1e30f;
#pragma unroll
    for (int e = 0; e < E_; ++e) { lg[e] = acc[e] + rb[e]; mx = fmaxf(mx, lg[e]); }
    float se = 0.f;
#pragma unroll
    for (int e = 0; e < E_; ++e) se += __expf(lg[e] - mx);
    float lse = mx + __logf(se);
    int best = 0; float bl = lg[0];
#pragma unroll
    for (int e = 1; e < E_; ++e) if (lg[e] > bl) { bl = lg[e]; best = e; }
#pragma unroll
    for (int e = 0; e < E_; ++e) atomicAdd(&tot_prob[e], __expf(lg[e] - mx) / se);
    atomicAdd(zsum, lse * lse);
    int pos = atomicAdd(&cnt[best], 1);
    list[best * T_ + pos] = tok;
  }
}

// ------------------------- gathered gate/up GEMM (bf16 [N,K] weights, async)
__global__ __launch_bounds__(256)
void moe_gateup(const bf16* __restrict__ x2, const bf16* __restrict__ gwT,
                const bf16* __restrict__ uwT, const int* __restrict__ cnt,
                const int* __restrict__ list, bf16* __restrict__ hbuf) {
  const int e = blockIdx.z, t0 = blockIdx.y * 64, n0 = blockIdx.x * 64;
  const int ce = cnt[e];
  if (t0 >= ce) return;
  __shared__ alignas(16) bf16 As[2][64 * 72];
  __shared__ alignas(16) bf16 Gs[2][64 * 72];
  __shared__ alignas(16) bf16 Us[2][64 * 72];
  __shared__ int rows[64];
  const int tid = threadIdx.x, w = tid >> 5, lane = tid & 31;
  const int wm = w >> 2, wn = w & 3;
  if (tid < 64) rows[tid] = (t0 + tid < ce) ? list[e * T_ + t0 + tid] : -1;
  __syncthreads();
  const bf16* gwe = gwT + (size_t)e * F_ * D_;   // [F, D] bf16
  const bf16* uwe = uwT + (size_t)e * F_ * D_;

  auto issue = [&](int k0, int buf) {
#pragma unroll
    for (int i = 0; i < 2; ++i) {
      int idx = tid + i * 256, r = idx >> 3, q = (idx & 7) * 8;
      int tk = rows[r]; if (tk < 0) tk = 0;      // finite dummy, masked later
      async_b128(&As[buf][r * 72 + q], &x2[(size_t)tk * D_ + k0 + q]);
      async_b128(&Gs[buf][r * 72 + q], &gwe[(size_t)(n0 + r) * D_ + k0 + q]);
      async_b128(&Us[buf][r * 72 + q], &uwe[(size_t)(n0 + r) * D_ + k0 + q]);
    }
  };

  issue(0, 0);
  v8f cg0 = {}, cg1 = {}, cu0 = {}, cu1 = {};
  const int nk = D_ >> 6;
  for (int t = 0; t < nk; ++t) {
    const int buf = t & 1;
    wait_async0();
    __syncthreads();
    if (t + 1 < nk) issue((t + 1) << 6, buf ^ 1);
#pragma unroll
    for (int kc = 0; kc < 2; ++kc) {
      v16bf bg = load_b_frag(&Gs[buf][(wn * 16) * 72 + kc * 32], 72);
      v16bf bu = load_b_frag(&Us[buf][(wn * 16) * 72 + kc * 32], 72);
      v16bf a0 = load_a_frag(&As[buf][(wm * 32) * 72 + kc * 32], 72);
      v16bf a1 = load_a_frag(&As[buf][(wm * 32 + 16) * 72 + kc * 32], 72);
      cg0 = wmma_bf16(a0, bg, cg0); cg1 = wmma_bf16(a1, bg, cg1);
      cu0 = wmma_bf16(a0, bu, cu0); cu1 = wmma_bf16(a1, bu, cu1);
    }
  }

  const int cn = n0 + wn * 16 + (lane & 15);
  const int rb0 = wm * 32 + ((lane >> 4) << 3);
#pragma unroll
  for (int r = 0; r < 8; ++r) {
    int tk0 = rows[rb0 + r];
    if (tk0 >= 0) {
      float g = cg0[r], u = cu0[r];
      hbuf[(size_t)tk0 * F_ + cn] = (bf16)(g / (1.f + __expf(-g)) * u);
    }
    int tk1 = rows[rb0 + 16 + r];
    if (tk1 >= 0) {
      float g = cg1[r], u = cu1[r];
      hbuf[(size_t)tk1 * F_ + cn] = (bf16)(g / (1.f + __expf(-g)) * u);
    }
  }
}

// --------------- gathered down GEMM, 128-token tile (bf16 [N,K] wt, async)
__global__ __launch_bounds__(256)
void moe_down(const bf16* __restrict__ hbuf, const bf16* __restrict__ dwT,
              const int* __restrict__ cnt, const int* __restrict__ list,
              const float* __restrict__ resid, float* __restrict__ out) {
  const int e = blockIdx.z, t0 = blockIdx.y * 128, n0 = blockIdx.x * 64;
  const int ce = cnt[e];
  if (t0 >= ce) return;
  __shared__ alignas(16) bf16 As[2][128 * 72];
  __shared__ alignas(16) bf16 Bs[2][64 * 72];
  __shared__ int rows[128];
  const int tid = threadIdx.x, w = tid >> 5, lane = tid & 31;
  const int wm = w >> 2, wn = w & 3;
  if (tid < 128) rows[tid] = (t0 + tid < ce) ? list[e * T_ + t0 + tid] : -1;
  __syncthreads();
  const bf16* dwe = dwT + (size_t)e * D_ * F_;   // [D, F] bf16

  auto issue = [&](int k0, int buf) {
#pragma unroll
    for (int i = 0; i < 4; ++i) {                // gathered A: 128 rows
      int idx = tid + i * 256, r = idx >> 3, q = (idx & 7) * 8;
      int tk = rows[r]; if (tk < 0) tk = 0;
      async_b128(&As[buf][r * 72 + q], &hbuf[(size_t)tk * F_ + k0 + q]);
    }
#pragma unroll
    for (int i = 0; i < 2; ++i) {
      int idx = tid + i * 256, r = idx >> 3, q = (idx & 7) * 8;
      async_b128(&Bs[buf][r * 72 + q], &dwe[(size_t)(n0 + r) * F_ + k0 + q]);
    }
  };

  issue(0, 0);
  v8f c[4] = {};
  const int nk = F_ >> 6;
  for (int t = 0; t < nk; ++t) {
    const int buf = t & 1;
    wait_async0();
    __syncthreads();
    if (t + 1 < nk) issue((t + 1) << 6, buf ^ 1);
#pragma unroll
    for (int kc = 0; kc < 2; ++kc) {
      v16bf bfr = load_b_frag(&Bs[buf][(wn * 16) * 72 + kc * 32], 72);
#pragma unroll
      for (int ms = 0; ms < 4; ++ms) {
        v16bf a = load_a_frag(&As[buf][(wm * 64 + ms * 16) * 72 + kc * 32], 72);
        c[ms] = wmma_bf16(a, bfr, c[ms]);
      }
    }
  }

  const int cn = n0 + wn * 16 + (lane & 15);
#pragma unroll
  for (int ms = 0; ms < 4; ++ms) {
    const int rb = wm * 64 + ms * 16 + ((lane >> 4) << 3);
#pragma unroll
    for (int r = 0; r < 8; ++r) {
      int tk = rows[rb + r];
      if (tk >= 0)
        out[(size_t)tk * D_ + cn] = resid[(size_t)tk * D_ + cn] + c[ms][r];
    }
  }
}

// -------------------------------------------------------------- misc small
__global__ void zero_counters(int* cnt, float* tp, float* zs) {
  int t = threadIdx.x;
  if (t < E_) { cnt[t] = 0; tp[t] = 0.f; }
  if (t == 0) zs[0] = 0.f;
}

__global__ void finalize_losses(const int* cnt, const float* tp,
                                const float* zs, float* tail) {
  if (threadIdx.x == 0) {
    float lb = 0.f;
    for (int e = 0; e < E_; ++e)
      lb += ((float)cnt[e] / (float)T_) * tp[e];
    tail[0] = (float)E_ * lb;
    tail[1] = (float)E_ * (zs[0] / (float)T_);
  }
}

// ---------------------------------------------------------------- launcher
extern "C" void kernel_launch(void* const* d_in, const int* in_sizes, int n_in,
                              void* d_out, int out_size, void* d_ws,
                              size_t ws_size, hipStream_t stream) {
  const float* hs  = (const float*)d_in[0];
  const float* ln1 = (const float*)d_in[1];
  const float* ln2 = (const float*)d_in[2];
  const float* wq  = (const float*)d_in[3];
  const float* wk  = (const float*)d_in[4];
  const float* wv  = (const float*)d_in[5];
  const float* wo  = (const float*)d_in[6];
  const float* gw  = (const float*)d_in[7];
  const float* uw  = (const float*)d_in[8];
  const float* dw  = (const float*)d_in[9];
  const float* rw  = (const float*)d_in[10];
  const float* rb  = (const float*)d_in[11];
  float* out = (float*)d_out;

  char* ws = (char*)d_ws;
  size_t off = 0;
  auto alloc = [&](size_t bytes) {
    void* p = ws + off;
    off = (off + bytes + 255) & ~(size_t)255;
    return p;
  };
  bf16*  x1   = (bf16*)alloc((size_t)T_ * D_ * 2);
  float* qf   = (float*)alloc((size_t)T_ * D_ * 4);
  float* kf   = (float*)alloc((size_t)T_ * D_ * 4);
  float* vf   = (float*)alloc((size_t)T_ * D_ * 4);
  bf16*  qb   = (bf16*)alloc((size_t)T_ * D_ * 2);
  bf16*  kb   = (bf16*)alloc((size_t)T_ * D_ * 2);
  bf16*  vb   = (bf16*)alloc((size_t)T_ * D_ * 2);
  bf16*  cb   = (bf16*)alloc((size_t)T_ * D_ * 2);
  float* hid2 = (float*)alloc((size_t)T_ * D_ * 4);
  bf16*  x2   = (bf16*)alloc((size_t)T_ * D_ * 2);
  bf16*  hbuf = (bf16*)alloc((size_t)T_ * F_ * 2);
  bf16*  wqb  = (bf16*)alloc((size_t)D_ * D_ * 2);
  bf16*  wkb  = (bf16*)alloc((size_t)D_ * D_ * 2);
  bf16*  wvb  = (bf16*)alloc((size_t)D_ * D_ * 2);
  bf16*  wob  = (bf16*)alloc((size_t)D_ * D_ * 2);
  bf16*  gwT  = (bf16*)alloc((size_t)E_ * F_ * D_ * 2);
  bf16*  uwT  = (bf16*)alloc((size_t)E_ * F_ * D_ * 2);
  bf16*  dwT  = (bf16*)alloc((size_t)E_ * D_ * F_ * 2);
  int*   cnt  = (int*)alloc(E_ * 4);
  int*   list = (int*)alloc((size_t)E_ * T_ * 4);
  float* tp   = (float*)alloc(E_ * 4);
  float* zs   = (float*)alloc(4);

  zero_counters<<<1, 32, 0, stream>>>(cnt, tp, zs);

  // one-time weight conversion (bf16 + layout normalization to [N,K])
  convert_f32_bf16<<<(D_ * D_) / 1024, 256, 0, stream>>>(wq, wqb, D_ * D_);
  convert_f32_bf16<<<(D_ * D_) / 1024, 256, 0, stream>>>(wk, wkb, D_ * D_);
  convert_f32_bf16<<<(D_ * D_) / 1024, 256, 0, stream>>>(wv, wvb, D_ * D_);
  convert_f32_bf16<<<(D_ * D_) / 1024, 256, 0, stream>>>(wo, wob, D_ * D_);
  transpose_f32_bf16<<<dim3(F_ / 64, D_ / 64, E_), 256, 0, stream>>>(gw, gwT,
                                                                     D_, F_);
  transpose_f32_bf16<<<dim3(F_ / 64, D_ / 64, E_), 256, 0, stream>>>(uw, uwT,
                                                                     D_, F_);
  transpose_f32_bf16<<<dim3(D_ / 64, F_ / 64, E_), 256, 0, stream>>>(dw, dwT,
                                                                     F_, D_);

  // attention block
  rmsnorm_bf16<<<T_, 256, 0, stream>>>(hs, ln1, x1);
  dim3 gqkv(D_ / 64, T_ / 128);
  gemm_nt_async<0><<<gqkv, 256, 0, stream>>>(x1, wqb, nullptr, qf, T_, D_, D_);
  gemm_nt_async<0><<<gqkv, 256, 0, stream>>>(x1, wkb, nullptr, kf, T_, D_, D_);
  gemm_nt_async<0><<<gqkv, 256, 0, stream>>>(x1, wvb, nullptr, vf, T_, D_, D_);
  rope_relayout<<<T_, 256, 0, stream>>>(qf, kf, vf, qb, kb, vb);
  flash_attn<<<dim3(S_ / 64, H_, B_), 128, 0, stream>>>(qb, kb, vb, cb);
  gemm_nt_async<1><<<gqkv, 256, 0, stream>>>(cb, wob, hs, hid2, T_, D_, D_);

  // MoE block
  rmsnorm_bf16<<<T_, 256, 0, stream>>>(hid2, ln2, x2);
  router_kernel<<<T_ / 8, 256, 0, stream>>>(x2, rw, rb, cnt, list, tp, zs);
  moe_gateup<<<dim3(F_ / 64, T_ / 64, E_), 256, 0, stream>>>(x2, gwT, uwT, cnt,
                                                             list, hbuf);
  moe_down<<<dim3(D_ / 64, T_ / 128, E_), 256, 0, stream>>>(hbuf, dwT, cnt,
                                                            list, hid2, out);
  finalize_losses<<<1, 32, 0, stream>>>(cnt, tp, zs, out + (size_t)T_ * D_);
}
